// LSTMBaselineModel_12962211299389
// MI455X (gfx1250) — matrix operand order repaired and checked by reading, not compiled, not run on hardware
//
#include <hip/hip_runtime.h>
#include <stdint.h>

#define SEQ   256
#define BATCH 256
#define IN_F  300
#define IN_P  320   // padded K for x / W_ih (10 k-tiles of 32)
#define HID   512
#define G4    2048  // 4*H
#define LF    256
#define BN_EPS 1e-5f

#define NK_X  (IN_P / 32)   // 10
#define NK_H  (HID / 32)    // 16

typedef __attribute__((ext_vector_type(16))) __bf16 v16bf;
typedef __attribute__((ext_vector_type(8)))  __bf16 v8bf;
typedef __attribute__((ext_vector_type(8)))  float  v8f;

// ---------------- helpers ----------------

__device__ __forceinline__ __bf16 f2bf(float f) {
  union { float f; uint32_t u; } in; in.f = f;
  uint32_t u = in.u;
  u += 0x7FFFu + ((u >> 16) & 1u);           // round-to-nearest-even
  union { uint16_t s; __bf16 b; } out; out.s = (uint16_t)(u >> 16);
  return out.b;
}

__device__ __forceinline__ float sigmoidf_fast(float x) {
  return 1.0f / (1.0f + __expf(-x));
}

// CDNA5 async DMA: global -> LDS, tracked by ASYNCcnt (no VGPR round-trip).
// the LDS address is taken from the VDST VGPR; generic LDS pointers carry the LDS offset
// in addr[31:0] (aperture rule), so truncation gives the VDST value.
__device__ __forceinline__ void async_ld_b128(uint32_t lds_off, const void* gaddr) {
  asm volatile("global_load_async_to_lds_b128 %0, %1, off"
               :: "v"(lds_off), "v"((uint64_t)(uintptr_t)gaddr) : "memory");
}
__device__ __forceinline__ void wait_async0() {
  asm volatile("s_wait_asynccnt 0x0" ::: "memory");
}

// ---------------- prep kernels ----------------

// fp32 [rows,K] -> bf16 [rows,Kp] zero-padded
__global__ void cvt_pad_kernel(const float* __restrict__ src, __bf16* __restrict__ dst,
                               int rows, int K, int Kp) {
  int total = rows * Kp;
  for (int i = blockIdx.x * blockDim.x + threadIdx.x; i < total;
       i += gridDim.x * blockDim.x) {
    int r = i / Kp, k = i - r * Kp;
    dst[i] = (k < K) ? f2bf(src[r * K + k]) : f2bf(0.0f);
  }
}

__global__ void bias_init_kernel(const float* __restrict__ b_ih,
                                 const float* __restrict__ b_hh,
                                 float* __restrict__ bias) {
  int i = blockIdx.x * blockDim.x + threadIdx.x;
  if (i < G4) bias[i] = b_ih[i] + b_hh[i];
}

__global__ void state_init_kernel(float* __restrict__ c, __bf16* __restrict__ hbf) {
  int i = blockIdx.x * blockDim.x + threadIdx.x;
  if (i < BATCH * HID) { c[i] = 0.0f; hbf[i] = f2bf(0.0f); }
}

// ---------------- WMMA GEMM building blocks ----------------
// Block = 256 threads (8 waves). Block tile: M=128 x N=64.
// Each wave owns a 16(M) x 64(N) strip: 1 A-fragment feeds 4 WMMAs per k-step.
// B tile (64 rows of W x 32 k) is staged to LDS by async DMA, double-buffered.

// stage one 64x32 bf16 tile: 256 threads x 16B async loads
__device__ __forceinline__ void stage_tile(const __bf16* __restrict__ W, int ldb,
                                           int n0, int k0, uint32_t lds_base) {
  int e   = threadIdx.x * 8;      // element index in tile (64*32 = 2048)
  int row = e >> 5;               // 0..63
  int col = e & 31;
  async_ld_b128(lds_base + (uint32_t)(e * 2),
                W + (size_t)(n0 + row) * ldb + k0 + col);
}

// gather the A fragment (16x32 bf16) in contiguous k order (same order as B):
// lanes 0-15: M=lane, K0..15; lanes 16-31: M=lane-16, K16..31
__device__ __forceinline__ v16bf load_a_frag(const __bf16* __restrict__ A, int lda,
                                             int m0, int k0, int lane) {
  int row = m0 + (lane & 15);
  const __bf16* pa = A + (size_t)row * lda + k0 + (lane >> 4) * 16;
  v8bf a_lo = *(const v8bf*)(pa);
  v8bf a_hi = *(const v8bf*)(pa + 8);
  v16bf a;
#pragma unroll
  for (int i = 0; i < 8; ++i) { a[i] = a_lo[i]; a[8 + i] = a_hi[i]; }
  __builtin_prefetch(pa + 32, 0, 0);   // global_prefetch_b8 for next k-tile
  return a;
}

// consume the current LDS tile: 4 B fragments -> 4 WMMAs
__device__ __forceinline__ void consume_tile(v8f acc[4], v16bf afrag,
                                             const __bf16* __restrict__ buf, int lane) {
  // B layout: lanes 0-15 K0..15 (N=lane), lanes 16-31 K16..31 (N=lane-16);
  // 32 contiguous bytes per lane.
  const int bo = (lane & 15) * 32 + (lane >> 4) * 16;
  v16bf b0 = *(const v16bf*)&buf[bo];
  v16bf b1 = *(const v16bf*)&buf[16 * 32 + bo];
  v16bf b2 = *(const v16bf*)&buf[32 * 32 + bo];
  v16bf b3 = *(const v16bf*)&buf[48 * 32 + bo];
  acc[0] = __builtin_amdgcn_wmma_f32_16x16x32_bf16(false, afrag, false, b0, (short)0, acc[0], false, false);
  acc[1] = __builtin_amdgcn_wmma_f32_16x16x32_bf16(false, afrag, false, b1, (short)0, acc[1], false, false);
  acc[2] = __builtin_amdgcn_wmma_f32_16x16x32_bf16(false, afrag, false, b2, (short)0, acc[2], false, false);
  acc[3] = __builtin_amdgcn_wmma_f32_16x16x32_bf16(false, afrag, false, b3, (short)0, acc[3], false, false);
}

// gates[B,4H] = x_t @ W_ih^T + h @ W_hh^T + (b_ih+b_hh)
// grid (2, 32), block 256
__global__ __launch_bounds__(256) void gemm_xh_kernel(
    const __bf16* __restrict__ xc_t, const __bf16* __restrict__ hbf,
    const __bf16* __restrict__ Wih, const __bf16* __restrict__ Whh,
    const float* __restrict__ bias, float* __restrict__ gates) {
  __shared__ __align__(16) __bf16 Bt[2][64 * 32];
  const int lane = threadIdx.x & 31;
  const int wave = threadIdx.x >> 5;
  const int m0 = blockIdx.x * 128 + wave * 16;
  const int n0 = blockIdx.y * 64;
  const uint32_t lb[2] = { (uint32_t)(uintptr_t)&Bt[0][0],
                           (uint32_t)(uintptr_t)&Bt[1][0] };

  v8f acc[4] = {};

  // prologue: stage x-tile 0
  stage_tile(Wih, IN_P, n0, 0, lb[0]);

  // ---- x segment: compile-time 10 k-steps, fully unrolled ----
#pragma unroll
  for (int s = 0; s < NK_X; ++s) {
    wait_async0();        // our slice of tile s is in LDS
    __syncthreads();      // all waves' slices in LDS; prev-tile reads retired
    if (s + 1 < NK_X) stage_tile(Wih, IN_P, n0, (s + 1) * 32, lb[(s + 1) & 1]);
    else              stage_tile(Whh, HID,  n0, 0,            lb[NK_X & 1]);
    v16bf afrag = load_a_frag(xc_t, IN_P, m0, s * 32, lane);
    consume_tile(acc, afrag, &Bt[s & 1][0], lane);
  }

  // ---- h segment: compile-time 16 k-steps, fully unrolled ----
#pragma unroll
  for (int s2 = 0; s2 < NK_H; ++s2) {
    const int s = NK_X + s2;
    wait_async0();
    __syncthreads();
    if (s2 + 1 < NK_H) stage_tile(Whh, HID, n0, (s2 + 1) * 32, lb[(s + 1) & 1]);
    v16bf afrag = load_a_frag(hbf, HID, m0, s2 * 32, lane);
    consume_tile(acc, afrag, &Bt[s & 1][0], lane);
  }

  // epilogue: C layout: lanes 0-15 rows M0..7, lanes 16-31 rows M8..15
  const int mrow = m0 + (lane >> 4) * 8;
#pragma unroll
  for (int j = 0; j < 4; ++j) {
    const int n = n0 + j * 16 + (lane & 15);
    const float bv = bias[n];
#pragma unroll
    for (int v = 0; v < 8; ++v)
      gates[(size_t)(mrow + v) * G4 + n] = acc[j][v] + bv;
  }
}

// y[B,L] = h @ W1^T + b1 ; grid (2, 4), block 256
__global__ __launch_bounds__(256) void gemm_head_kernel(
    const __bf16* __restrict__ hbf, const __bf16* __restrict__ W1b,
    const float* __restrict__ b1, float* __restrict__ y) {
  __shared__ __align__(16) __bf16 Bt[2][64 * 32];
  const int lane = threadIdx.x & 31;
  const int wave = threadIdx.x >> 5;
  const int m0 = blockIdx.x * 128 + wave * 16;
  const int n0 = blockIdx.y * 64;
  const uint32_t lb[2] = { (uint32_t)(uintptr_t)&Bt[0][0],
                           (uint32_t)(uintptr_t)&Bt[1][0] };

  v8f acc[4] = {};
  stage_tile(W1b, HID, n0, 0, lb[0]);

#pragma unroll
  for (int s = 0; s < NK_H; ++s) {
    wait_async0();
    __syncthreads();
    if (s + 1 < NK_H) stage_tile(W1b, HID, n0, (s + 1) * 32, lb[(s + 1) & 1]);
    v16bf afrag = load_a_frag(hbf, HID, m0, s * 32, lane);
    consume_tile(acc, afrag, &Bt[s & 1][0], lane);
  }

  const int mrow = m0 + (lane >> 4) * 8;
#pragma unroll
  for (int j = 0; j < 4; ++j) {
    const int n = n0 + j * 16 + (lane & 15);
    const float bv = b1[n];
#pragma unroll
    for (int v = 0; v < 8; ++v)
      y[(size_t)(mrow + v) * LF + n] = acc[j][v] + bv;
  }
}

// ---------------- LSTM cell elementwise ----------------
// PyTorch gate order i,f,g,o ; 512 blocks x 256 threads = B*H
__global__ void cell_kernel(const float* __restrict__ gates,
                            float* __restrict__ c, __bf16* __restrict__ hbf) {
  int idx = blockIdx.x * blockDim.x + threadIdx.x;   // < BATCH*HID
  int b = idx >> 9, j = idx & 511;
  const float* g = gates + (size_t)b * G4;
  float i_ = sigmoidf_fast(g[j]);
  float f_ = sigmoidf_fast(g[HID + j]);
  float g_ = tanhf(g[2 * HID + j]);
  float o_ = sigmoidf_fast(g[3 * HID + j]);
  float cn = f_ * c[idx] + i_ * g_;
  c[idx] = cn;
  hbf[idx] = f2bf(o_ * tanhf(cn));
}

// ---------------- BatchNorm stats (train-mode, over batch) ----------------
__global__ void bn_stats_kernel(const float* __restrict__ y,
                                const float* __restrict__ gamma,
                                const float* __restrict__ beta,
                                float* __restrict__ scale, float* __restrict__ shift) {
  int j = threadIdx.x;   // feature, 256 threads / 1 block
  float s = 0.0f, sq = 0.0f;
  for (int b = 0; b < BATCH; ++b) {
    float v = y[b * LF + j];
    s += v; sq += v * v;
  }
  float mu  = s * (1.0f / BATCH);
  float var = sq * (1.0f / BATCH) - mu * mu;
  float sc  = gamma[j] * rsqrtf(var + BN_EPS);
  scale[j] = sc;
  shift[j] = beta[j] - mu * sc;
}

// ---------------- final: relu(BN(y)) @ W2^T + b2, 3*sigmoid ----------------
__global__ void final_kernel(const float* __restrict__ y,
                             const float* __restrict__ scale,
                             const float* __restrict__ shift,
                             const float* __restrict__ W2,
                             const float* __restrict__ b2,
                             float* __restrict__ out) {
  __shared__ float red[256];
  int b = blockIdx.x, l = threadIdx.x;
  float v = y[b * LF + l] * scale[l] + shift[l];
  v = fmaxf(v, 0.0f) * W2[l];
  red[l] = v;
  __syncthreads();
  for (int s = 128; s > 0; s >>= 1) {
    if (l < s) red[l] += red[l + s];
    __syncthreads();
  }
  if (l == 0) out[b] = 3.0f * sigmoidf_fast(red[0] + b2[0]);
}

// ---------------- launch ----------------

extern "C" void kernel_launch(void* const* d_in, const int* in_sizes, int n_in,
                              void* d_out, int out_size, void* d_ws, size_t ws_size,
                              hipStream_t stream) {
  const float* x     = (const float*)d_in[0];
  const float* W_ih  = (const float*)d_in[1];
  const float* W_hh  = (const float*)d_in[2];
  const float* b_ih  = (const float*)d_in[3];
  const float* b_hh  = (const float*)d_in[4];
  const float* W1    = (const float*)d_in[5];
  const float* b1    = (const float*)d_in[6];
  const float* gamma = (const float*)d_in[7];
  const float* beta  = (const float*)d_in[8];
  const float* W2    = (const float*)d_in[9];
  const float* b2    = (const float*)d_in[10];
  float* out = (float*)d_out;

  char* ws = (char*)d_ws;
  size_t off = 0;
  auto alloc = [&](size_t bytes) -> void* {
    void* p = ws + off;
    off = (off + bytes + 255) & ~(size_t)255;
    return p;
  };
  __bf16* xc   = (__bf16*)alloc((size_t)SEQ * BATCH * IN_P * 2);  // ~40 MB
  __bf16* Wihb = (__bf16*)alloc((size_t)G4 * IN_P * 2);
  __bf16* Whhb = (__bf16*)alloc((size_t)G4 * HID * 2);
  __bf16* W1b  = (__bf16*)alloc((size_t)LF * HID * 2);
  float*  bias = (float*) alloc((size_t)G4 * 4);
  float*  gate = (float*) alloc((size_t)BATCH * G4 * 4);
  float*  cst  = (float*) alloc((size_t)BATCH * HID * 4);
  __bf16* hbf  = (__bf16*)alloc((size_t)BATCH * HID * 2);
  float*  yb   = (float*) alloc((size_t)BATCH * LF * 4);
  float*  scl  = (float*) alloc((size_t)LF * 4);
  float*  shf  = (float*) alloc((size_t)LF * 4);

  // one-time prep
  cvt_pad_kernel<<<4096, 256, 0, stream>>>(x,    xc,   SEQ * BATCH, IN_F, IN_P);
  cvt_pad_kernel<<<1024, 256, 0, stream>>>(W_ih, Wihb, G4,          IN_F, IN_P);
  cvt_pad_kernel<<<1024, 256, 0, stream>>>(W_hh, Whhb, G4,          HID,  HID);
  cvt_pad_kernel<<<256,  256, 0, stream>>>(W1,   W1b,  LF,          HID,  HID);
  bias_init_kernel<<<8,   256, 0, stream>>>(b_ih, b_hh, bias);
  state_init_kernel<<<512, 256, 0, stream>>>(cst, hbf);

  // sequential recurrence
  for (int t = 0; t < SEQ; ++t) {
    const __bf16* xt = xc + (size_t)t * BATCH * IN_P;
    gemm_xh_kernel<<<dim3(2, 32), 256, 0, stream>>>(xt, hbf, Wihb, Whhb, bias, gate);
    cell_kernel<<<512, 256, 0, stream>>>(gate, cst, hbf);
  }

  // head
  gemm_head_kernel<<<dim3(2, 4), 256, 0, stream>>>(hbf, W1b, b1, yb);
  bn_stats_kernel<<<1, 256, 0, stream>>>(yb, gamma, beta, scl, shf);
  final_kernel<<<256, 256, 0, stream>>>(yb, scl, shf, W2, b2, out);
}
